// MAB_2_83021717832762
// MI455X (gfx1250) — matrix operand order repaired and checked by reading, not compiled, not run on hardware
//
#include <hip/hip_runtime.h>
#include <hip/hip_bf16.h>
#include <stdint.h>

// ---------------- problem constants (match reference) ----------------
constexpr int kB   = 2;
constexpr int kDIM = 1024;
constexpr int kH   = 16;
constexpr int kDK  = 64;
constexpr int kSQ  = 2048;
constexpr int kSK  = 2048;
constexpr int kD   = kH * kDK;   // 1024

// ---------------- vector types for WMMA ----------------
typedef _Float16 v8h  __attribute__((ext_vector_type(8)));
typedef _Float16 v16h __attribute__((ext_vector_type(16)));
typedef float    v8f  __attribute__((ext_vector_type(8)));

static __device__ __forceinline__ v16h cat16(v8h lo, v8h hi) {
    return __builtin_shufflevector(lo, hi, 0,1,2,3,4,5,6,7,8,9,10,11,12,13,14,15);
}

static __device__ __forceinline__ v8f wmma_f16(v16h a, v16h b, v8f c) {
    // D = A(16x32 f16) * B(32x16 f16) + C(16x16 f32)
    return __builtin_amdgcn_wmma_f32_16x16x32_f16(
        /*neg_a=*/false, a, /*neg_b=*/false, b,
        /*c_mod=*/(short)0, c, /*reuse_a=*/false, /*reuse_b=*/false);
}

// wave-local LDS write->read fence (per-wave regions, no block barrier needed)
static __device__ __forceinline__ void wave_lds_fence() {
    asm volatile("s_wait_dscnt 0x0" ::: "memory");
}

// ---------------- plain f32 -> f16 conversion (weights) ----------------
__global__ void cvt_f32_to_f16(const float* __restrict__ in,
                               _Float16* __restrict__ out, size_t n) {
    size_t i = (size_t)blockIdx.x * blockDim.x + threadIdx.x;
    size_t stride = (size_t)gridDim.x * blockDim.x;
    for (; i < n; i += stride) out[i] = (_Float16)in[i];
}

// ---------------- f32 [C][S] -> f16 [S][C] tiled transpose+convert ----------------
__global__ __launch_bounds__(256)
void cvt_transpose_f16(const float* __restrict__ in,   // [z][C][S]
                       _Float16* __restrict__ out,     // [z][S][C]
                       int C, int S) {
    __shared__ float tile[32][33];
    const int z = blockIdx.z;
    const float* inz = in + (size_t)z * C * S;
    _Float16* outz = out + (size_t)z * S * C;

    const int s0 = blockIdx.x * 32;
    const int c0 = blockIdx.y * 32;
    const int tx = threadIdx.x;      // 0..31
    const int ty = threadIdx.y;      // 0..7

    #pragma unroll
    for (int i = 0; i < 4; ++i)
        tile[ty + i * 8][tx] = inz[(size_t)(c0 + ty + i * 8) * S + s0 + tx];
    __syncthreads();
    #pragma unroll
    for (int i = 0; i < 4; ++i)
        outz[(size_t)(s0 + ty + i * 8) * C + c0 + tx] = (_Float16)tile[tx][ty + i * 8];
}

// ---------------- barrier-free direct WMMA GEMM ----------------
// C[M,N] = act(A[M,K] * Bt[N,K]^T + bias[M]) (+ resid[M,N])
// A : f16 row-major [M][K]       (weights, shared across batch)
// Bt: f16 "transposed" [N][K]    (sequence-major activations), per-batch strideB
// flags: 1=ReLU, 2=f16 store [m][n], 4=f16 store transposed [n][m],
//        8=f32 store [m][n], 16=add f32 residual [m][n]
#define F_RELU 1
#define F_F16N 2
#define F_F16T 4
#define F_F32N 8
#define F_RES  16

__global__ __launch_bounds__(256)
void gemm_wmma_kernel(const _Float16* __restrict__ A,
                      const _Float16* __restrict__ Bt,
                      const float* __restrict__ bias,
                      const float* __restrict__ resid,
                      _Float16* __restrict__ outh,
                      float* __restrict__ outf,
                      int M, int N, int K,
                      long strideB, long strideC, long strideR,
                      int flags)
{
    const int t     = threadIdx.x;
    const int lane  = t & 31;
    const int wave  = t >> 5;       // 0..7
    const int wm    = wave & 3;     // 4 m-waves x 32 = 128 M
    const int wn    = wave >> 2;    // 2 n-waves x 64 = 128 N
    const int z     = blockIdx.z;
    const int myN   = blockIdx.x * 128 + wn * 64;
    const int myM   = blockIdx.y * 128 + wm * 32;

    const int l16   = lane & 15;
    const int lhalf = lane >> 4;
    const int c0    = lhalf * 8;     // A-frag chunk offsets c0 / 16+c0
    const int bk0   = lhalf * 16;    // unused; B-frag uses c0 / 16+c0 like A

    const _Float16* Bz = Bt + (size_t)z * strideB;

    const v8f zf = {0.f,0.f,0.f,0.f,0.f,0.f,0.f,0.f};
    v8f acc[2][4];
    #pragma unroll
    for (int ms = 0; ms < 2; ++ms)
        #pragma unroll
        for (int ns = 0; ns < 4; ++ns) acc[ms][ns] = zf;

    const _Float16* arow0 = &A [(size_t)(myM + l16)       * K];
    const _Float16* arow1 = &A [(size_t)(myM + 16 + l16)  * K];

    for (int kb = 0; kb < K; kb += 32) {
        v16h af[2];
        af[0] = cat16(*reinterpret_cast<const v8h*>(arow0 + kb + c0),
                      *reinterpret_cast<const v8h*>(arow0 + kb + 16 + c0));
        af[1] = cat16(*reinterpret_cast<const v8h*>(arow1 + kb + c0),
                      *reinterpret_cast<const v8h*>(arow1 + kb + 16 + c0));

        if (kb + 32 < K)
            __builtin_prefetch(arow0 + kb + 32, 0, 3);

        #pragma unroll
        for (int ns = 0; ns < 4; ++ns) {
            const _Float16* brow = &Bz[(size_t)(myN + ns * 16 + l16) * K + kb + c0];
            v16h bf = cat16(*reinterpret_cast<const v8h*>(brow),
                            *reinterpret_cast<const v8h*>(brow + 16));
            acc[0][ns] = wmma_f16(af[0], bf, acc[0][ns]);
            acc[1][ns] = wmma_f16(af[1], bf, acc[1][ns]);
        }
    }

    // ---- epilogue ----
    #pragma unroll
    for (int ms = 0; ms < 2; ++ms) {
        const int mbase = myM + ms * 16 + lhalf * 8;
        v8f bv = *reinterpret_cast<const v8f*>(&bias[mbase]);   // 8 consecutive m
        #pragma unroll
        for (int ns = 0; ns < 4; ++ns) {
            const int n = myN + ns * 16 + l16;
            float tmp[8];
            #pragma unroll
            for (int r = 0; r < 8; ++r) {
                float v = acc[ms][ns][r] + bv[r];
                if (flags & F_RELU) v = v > 0.f ? v : 0.f;
                if (flags & F_RES)
                    v += resid[(size_t)z * strideR + (size_t)(mbase + r) * N + n];
                tmp[r] = v;
            }
            if (flags & F_F16T) {     // [n][m]: m contiguous -> one 16B store
                v8h hv;
                #pragma unroll
                for (int r = 0; r < 8; ++r) hv[r] = (_Float16)tmp[r];
                *reinterpret_cast<v8h*>(
                    &outh[(size_t)z * strideC + (size_t)n * M + mbase]) = hv;
            }
            if (flags & F_F16N) {
                #pragma unroll
                for (int r = 0; r < 8; ++r)
                    outh[(size_t)z * strideC + (size_t)(mbase + r) * N + n] = (_Float16)tmp[r];
            }
            if (flags & F_F32N) {
                #pragma unroll
                for (int r = 0; r < 8; ++r)
                    outf[(size_t)z * strideC + (size_t)(mbase + r) * N + n] = tmp[r];
            }
        }
    }
}

// ---------------- sigmoid-renormalized attention ----------------
// qf: [B][SQ][D], kf: [B][SK][D], vf: [B][D][SK], mask: [B][SK]
// oatt: [B][SQ][D]  (sequence-major for the next GEMM)
__global__ __launch_bounds__(128)
void attn_sigmoid_kernel(const _Float16* __restrict__ qf,
                         const _Float16* __restrict__ kf,
                         const _Float16* __restrict__ vf,
                         const float* __restrict__ mask,
                         _Float16* __restrict__ oatt)
{
    __shared__ _Float16 P[4][16][40];   // per-wave sigmoid(P) tile [q=16][k=32]

    const int t     = threadIdx.x;
    const int lane  = t & 31;
    const int wave  = t >> 5;           // 4 independent waves
    const int l16   = lane & 15;
    const int lhalf = lane >> 4;
    const int c0    = lhalf * 8;

    const int h     = blockIdx.y;
    const int b     = blockIdx.z;
    const int qbase = blockIdx.x * 64 + wave * 16;
    const int dhead = h * kDK;

    const _Float16* qb = qf + (size_t)b * kSQ * kD;
    const _Float16* kb = kf + (size_t)b * kSK * kD;
    const _Float16* vb = vf + (size_t)b * kD * kSK;
    const float*    mb = mask + (size_t)b * kSK;
    _Float16*       ob = oatt + (size_t)b * kSQ * kD;

    // q A-fragments for the two K=32 chunks of DK=64 (loaded once)
    v16h aq[2];
    {
        const _Float16* qrow = &qb[(size_t)(qbase + l16) * kD + dhead];
        #pragma unroll
        for (int dk = 0; dk < 2; ++dk)
            aq[dk] = cat16(*reinterpret_cast<const v8h*>(qrow + dk * 32 + c0),
                           *reinterpret_cast<const v8h*>(qrow + dk * 32 + 16 + c0));
    }

    v16h ones;
    #pragma unroll
    for (int i = 0; i < 16; ++i) ones[i] = (_Float16)1.0f;

    const v8f zf = {0.f,0.f,0.f,0.f,0.f,0.f,0.f,0.f};
    v8f oacc[4]; oacc[0] = zf; oacc[1] = zf; oacc[2] = zf; oacc[3] = zf;
    v8f sacc = zf;

    const float scale = 0.125f;   // 1/sqrt(64)

    for (int kt = 0; kt < kSK; kt += 32) {
        // ---- logits 16x32 tile ----
        v8f p[2]; p[0] = zf; p[1] = zf;
        #pragma unroll
        for (int ns = 0; ns < 2; ++ns) {
            #pragma unroll
            for (int dk = 0; dk < 2; ++dk) {
                const _Float16* kr =
                    &kb[(size_t)(kt + ns * 16 + l16) * kD + dhead + dk * 32 + c0];
                v16h bk = cat16(*reinterpret_cast<const v8h*>(kr),
                                *reinterpret_cast<const v8h*>(kr + 16));
                p[ns] = wmma_f16(aq[dk], bk, p[ns]);
            }
        }

        // ---- mask + sigmoid, reshape C-frag -> A-frag via per-wave LDS ----
        #pragma unroll
        for (int ns = 0; ns < 2; ++ns) {
            const float madd = (1.0f - mb[kt + ns * 16 + l16]) * 10000.0f;
            #pragma unroll
            for (int r = 0; r < 8; ++r) {
                float x = p[ns][r] * scale - madd;
                float s = 1.0f / (1.0f + __expf(-x));
                P[wave][lhalf * 8 + r][ns * 16 + l16] = (_Float16)s;
            }
        }
        wave_lds_fence();   // per-wave region: s_wait_dscnt 0, no block barrier

        v16h pa;
        {
            const _Float16* pr = &P[wave][l16][0];
            pa = cat16(*reinterpret_cast<const v8h*>(pr + c0),
                       *reinterpret_cast<const v8h*>(pr + 16 + c0));
        }

        // ---- O += P * V ----
        #pragma unroll
        for (int ns = 0; ns < 4; ++ns) {
            const _Float16* vr =
                &vb[(size_t)(dhead + ns * 16 + l16) * kSK + kt + c0];
            v16h bvv = cat16(*reinterpret_cast<const v8h*>(vr),
                             *reinterpret_cast<const v8h*>(vr + 16));
            oacc[ns] = wmma_f16(pa, bvv, oacc[ns]);
        }
        // ---- row-sum via WMMA with all-ones B: same C layout as O ----
        sacc = wmma_f16(pa, ones, sacc);
        wave_lds_fence();   // pa reads done before next iteration overwrites P
    }

    // ---- renormalize, store sequence-major [SQ][D] ----
    #pragma unroll
    for (int ns = 0; ns < 4; ++ns) {
        const int d = dhead + ns * 16 + l16;
        #pragma unroll
        for (int r = 0; r < 8; ++r) {
            const int q = qbase + lhalf * 8 + r;
            ob[(size_t)q * kD + d] = (_Float16)(oacc[ns][r] / sacc[r]);
        }
    }
}

// ---------------- host orchestration ----------------
extern "C" void kernel_launch(void* const* d_in, const int* in_sizes, int n_in,
                              void* d_out, int out_size, void* d_ws, size_t ws_size,
                              hipStream_t stream) {
    const float* Q    = (const float*)d_in[0];
    const float* Kin  = (const float*)d_in[1];
    const float* mask = (const float*)d_in[2];
    const float* wq   = (const float*)d_in[3];
    const float* bq   = (const float*)d_in[4];
    const float* wk   = (const float*)d_in[5];
    const float* bk   = (const float*)d_in[6];
    const float* wv   = (const float*)d_in[7];
    const float* bv   = (const float*)d_in[8];
    const float* wp   = (const float*)d_in[9];
    const float* bp   = (const float*)d_in[10];
    const float* w1   = (const float*)d_in[11];
    const float* b1   = (const float*)d_in[12];
    const float* w2   = (const float*)d_in[13];
    const float* b2   = (const float*)d_in[14];
    float* out = (float*)d_out;

    const size_t NACT = (size_t)kB * kDIM * kSQ;      // 4,194,304
    const size_t NW   = (size_t)kD * kDIM;            // 1,048,576
    const size_t NW2  = (size_t)2 * kDIM * kDIM;      // 2,097,152

    char* ws = (char*)d_ws;
    size_t off = 0;
    auto alloc = [&](size_t bytes) -> char* {
        char* p = ws + off;
        off += (bytes + 255) & ~(size_t)255;
        return p;
    };

    _Float16* wqh = (_Float16*)alloc(NW  * 2);
    _Float16* wkh = (_Float16*)alloc(NW  * 2);
    _Float16* wvh = (_Float16*)alloc(NW  * 2);
    _Float16* wph = (_Float16*)alloc(NW  * 2);
    _Float16* w1h = (_Float16*)alloc(NW2 * 2);
    _Float16* w2h = (_Float16*)alloc(NW2 * 2);
    _Float16* Qt  = (_Float16*)alloc(NACT * 2);   // [B][SQ][DIM]; reused as oatt
    _Float16* Kt  = (_Float16*)alloc(NACT * 2);   // [B][SK][DIM]; reused as oph
    _Float16* qf  = (_Float16*)alloc(NACT * 2);   // [B][SQ][D]
    _Float16* kf  = (_Float16*)alloc(NACT * 2);   // [B][SK][D]
    _Float16* vf  = (_Float16*)alloc(NACT * 2);   // [B][D][SK]
    float*    opf = (float*)   alloc(NACT * 4);   // o_p f32 [B][DIM][SQ]
    _Float16* hh  = (_Float16*)alloc((size_t)kB * kSQ * 2 * kDIM * 2);  // [B][SQ][2DIM]
    (void)ws_size; (void)in_sizes; (void)n_in; (void)out_size;

    // ---- 1. input transpose+convert and weight conversion ----
    cvt_transpose_f16<<<dim3(kSQ/32, kDIM/32, kB), dim3(32,8), 0, stream>>>(Q,   Qt, kDIM, kSQ);
    cvt_transpose_f16<<<dim3(kSK/32, kDIM/32, kB), dim3(32,8), 0, stream>>>(Kin, Kt, kDIM, kSK);
    {
        dim3 blk(256);
        cvt_f32_to_f16<<<dim3(512),  blk, 0, stream>>>(wq, wqh, NW);
        cvt_f32_to_f16<<<dim3(512),  blk, 0, stream>>>(wk, wkh, NW);
        cvt_f32_to_f16<<<dim3(512),  blk, 0, stream>>>(wv, wvh, NW);
        cvt_f32_to_f16<<<dim3(512),  blk, 0, stream>>>(wp, wph, NW);
        cvt_f32_to_f16<<<dim3(1024), blk, 0, stream>>>(w1, w1h, NW2);
        cvt_f32_to_f16<<<dim3(1024), blk, 0, stream>>>(w2, w2h, NW2);
    }

    const dim3 gblk(256);

    // ---- 2. q = wq*Q + bq -> qf [B][SQ][D] ----
    gemm_wmma_kernel<<<dim3(kSQ/128, kD/128, kB), gblk, 0, stream>>>(
        wqh, Qt, bq, nullptr, qf, nullptr,
        kD, kSQ, kDIM, (long)kSQ*kDIM, (long)kSQ*kD, 0, F_F16T);

    // ---- 3. k = wk*K + bk -> kf [B][SK][D] ----
    gemm_wmma_kernel<<<dim3(kSK/128, kD/128, kB), gblk, 0, stream>>>(
        wkh, Kt, bk, nullptr, kf, nullptr,
        kD, kSK, kDIM, (long)kSK*kDIM, (long)kSK*kD, 0, F_F16T);

    // ---- 4. v = wv*K + bv -> vf [B][D][SK] ----
    gemm_wmma_kernel<<<dim3(kSK/128, kD/128, kB), gblk, 0, stream>>>(
        wvh, Kt, bv, nullptr, vf, nullptr,
        kD, kSK, kDIM, (long)kSK*kDIM, (long)kD*kSK, 0, F_F16N);

    // ---- 5. attention -> oatt [B][SQ][D] (reuses Qt) ----
    _Float16* oatt = Qt;
    attn_sigmoid_kernel<<<dim3(kSQ/64, kH, kB), dim3(128), 0, stream>>>(
        qf, kf, vf, mask, oatt);

    // ---- 6. o = wp*oatt + bp + Q : f16 [SQ][DIM] (reuses Kt) + f32 [DIM][SQ] ----
    _Float16* oph = Kt;
    gemm_wmma_kernel<<<dim3(kSQ/128, kDIM/128, kB), gblk, 0, stream>>>(
        wph, oatt, bp, Q, oph, opf,
        kDIM, kSQ, kD, (long)kSQ*kD, (long)kDIM*kSQ, (long)kDIM*kSQ,
        F_F16T | F_F32N | F_RES);

    // ---- 7. h = relu(w1*o + b1) -> hh [B][SQ][2DIM] ----
    gemm_wmma_kernel<<<dim3(kSQ/128, (2*kDIM)/128, kB), gblk, 0, stream>>>(
        w1h, oph, b1, nullptr, hh, nullptr,
        2*kDIM, kSQ, kDIM, (long)kSQ*kDIM, (long)kSQ*2*kDIM, 0,
        F_RELU | F_F16T);

    // ---- 8. out = w2*h + b2 + o -> d_out f32 [B][DIM][SQ] ----
    gemm_wmma_kernel<<<dim3(kSQ/128, kDIM/128, kB), gblk, 0, stream>>>(
        w2h, hh, b2, opf, nullptr, out,
        kDIM, kSQ, 2*kDIM, (long)kSQ*2*kDIM, (long)kDIM*kSQ, (long)kDIM*kSQ,
        F_F32N | F_RES);
}
